// NMSModel_687194767746
// MI455X (gfx1250) — 8-bit matrix operand order repaired and checked by reading, compiled, not run on hardware
//
#include <hip/hip_runtime.h>
#include <hip/hip_bf16.h>
#include <stdint.h>


#define NCC      80
#define AA       8400
#define BB       32
#define TOPK     1500
#define MAX_DET  300
#define PAD_N    1536
#define SORTN    2048
#define CONF_TH  0.25f
#define IOU_TH   0.45f
#define MULT     (1.0f / 80.0f)
#define BOX_SCALE (MULT / 640.0f)

typedef int v8i __attribute__((ext_vector_type(8)));

__device__ inline uint32_t scoreToKey(float f) {
    uint32_t b = __float_as_uint(f);
    return (b & 0x80000000u) ? ~b : (b | 0x80000000u);
}
__device__ inline float keyToScore(uint32_t u) {
    uint32_t b = (u & 0x80000000u) ? (u ^ 0x80000000u) : ~u;
    return __uint_as_float(b);
}

__device__ inline bool iouOver(float4 p, float ap, float4 q, float aq) {
    float lx = fmaxf(p.x, q.x), ly = fmaxf(p.y, q.y);
    float rx = fminf(p.z, q.z), ry = fminf(p.w, q.w);
    float iw = fmaxf(rx - lx, 0.0f), ih = fmaxf(ry - ly, 0.0f);
    float inter = iw * ih;
    return inter > IOU_TH * (ap + aq - inter + 1e-7f);
}

__global__ __launch_bounds__(256) void score_kernel(const float* __restrict__ pred,
                                                    uint32_t* __restrict__ keys,
                                                    uint32_t* __restrict__ cls) {
    int b = blockIdx.y;
    int a = blockIdx.x * blockDim.x + threadIdx.x;
    if (a >= AA) return;
    const float* p = pred + (size_t)b * (NCC + 4) * AA + (size_t)4 * AA + a;
    float best = -1.0f;
    int bc = 0;
#pragma unroll 4
    for (int c = 0; c < NCC; ++c) {
        if ((c & 15) == 0 && (c + 16) < NCC)
            __builtin_prefetch(p + (size_t)(c + 16) * AA, 0, 0);
        float v = p[(size_t)c * AA];
        if (v > best) { best = v; bc = c; }
    }
    float msc = (best > CONF_TH) ? best : -1.0f;
    keys[(size_t)b * AA + a] = scoreToKey(msc);
    cls [(size_t)b * AA + a] = (uint32_t)bc;
}

__global__ __launch_bounds__(1024) void topk_kernel(const uint32_t* __restrict__ keys,
                                                    uint32_t* __restrict__ topkKey,
                                                    uint32_t* __restrict__ topkIdx) {
    __shared__ __align__(16) uint32_t sKeys[AA];
    __shared__ uint32_t sHist[2048];
    __shared__ unsigned long long sSort[SORTN];
    __shared__ uint32_t sState[4];

    const int b = blockIdx.x, tid = threadIdx.x, nt = blockDim.x;

    const uint32_t* src = keys + (size_t)b * AA;
    uint32_t ldsBase = (uint32_t)(size_t)(&sKeys[0]);
    for (int i = tid; i < AA / 4; i += nt) {
        uint32_t la = ldsBase + (uint32_t)(i * 16);
        uint64_t ga = (uint64_t)(size_t)(src + (size_t)i * 4);
        asm volatile("global_load_async_to_lds_b128 %0, %1, off"
                     :: "v"(la), "v"(ga) : "memory");
    }
    asm volatile("s_wait_asynccnt 0" ::: "memory");
    if (tid == 0) { sState[0] = 0u; sState[1] = 0u; sState[2] = 0u; sState[3] = 0u; }
    __syncthreads();

    const int shifts[3]  = {21, 10, 0};
    const int nbuck[3]   = {2048, 2048, 1024};
    const int mshift[3]  = {0, 21, 10};
    for (int r = 0; r < 3; ++r) {
        for (int d = tid; d < 2048; d += nt) sHist[d] = 0u;
        __syncthreads();
        uint32_t pref = sState[0];
        for (int a = tid; a < AA; a += nt) {
            uint32_t u = sKeys[a];
            bool match = (r == 0) || ((u >> mshift[r]) == pref);
            if (match) atomicAdd(&sHist[(u >> shifts[r]) & (uint32_t)(nbuck[r] - 1)], 1u);
        }
        __syncthreads();
        if (tid == 0) {
            uint32_t countAbove = sState[1];
            int rem = TOPK - (int)countAbove;
            int chosen = 0;
            for (int d = nbuck[r] - 1; d >= 0; --d) {
                uint32_t h = sHist[d];
                if ((int)h >= rem) { chosen = d; break; }
                rem -= (int)h; countAbove += h;
            }
            sState[1] = countAbove;
            if      (r == 0) sState[0] = (uint32_t)chosen;
            else if (r == 1) sState[0] = (sState[0] << 11) | (uint32_t)chosen;
            else             sState[2] = (sState[0] << 10) | (uint32_t)chosen;
        }
        __syncthreads();
    }
    const uint32_t ustar = sState[2];

    for (int a = tid; a < AA; a += nt) {
        uint32_t u = sKeys[a];
        if (u >= ustar) {
            uint32_t p = atomicAdd(&sState[3], 1u);
            if (p < SORTN)
                sSort[p] = ((unsigned long long)u << 32) |
                           (unsigned long long)(0xFFFFFFFFu - (uint32_t)a);
        }
    }
    __syncthreads();
    uint32_t cnt = sState[3]; if (cnt > SORTN) cnt = SORTN;
    for (int p = tid; p < SORTN; p += nt) if (p >= (int)cnt) sSort[p] = 0ull;
    __syncthreads();

    for (int k = 2; k <= SORTN; k <<= 1) {
        for (int j = k >> 1; j > 0; j >>= 1) {
            for (int e = tid; e < SORTN; e += nt) {
                int l = e ^ j;
                if (l > e) {
                    unsigned long long x = sSort[e], y = sSort[l];
                    bool sw = ((e & k) == 0) ? (x < y) : (x > y);
                    if (sw) { sSort[e] = y; sSort[l] = x; }
                }
            }
            __syncthreads();
        }
    }

    for (int rI = tid; rI < TOPK; rI += nt) {
        unsigned long long kv = sSort[rI];
        topkKey[(size_t)b * TOPK + rI] = (uint32_t)(kv >> 32);
        topkIdx[(size_t)b * TOPK + rI] = 0xFFFFFFFFu - (uint32_t)kv;
    }
}

__global__ __launch_bounds__(1024) void nms_kernel(const float* __restrict__ pred,
                                                   const uint32_t* __restrict__ cls,
                                                   const uint32_t* __restrict__ topkKey,
                                                   const uint32_t* __restrict__ topkIdx,
                                                   float* __restrict__ out) {
    __shared__ float4 sNms[PAD_N];
    __shared__ float  sArea[PAD_N];
    __shared__ uint32_t sSupp[PAD_N / 32];
    __shared__ uint32_t sValid[PAD_N / 32];
    __shared__ unsigned char sKeep[64];
    __shared__ uint32_t sKeepW[PAD_N / 32];
    __shared__ uint32_t sPref[PAD_N / 32 + 1];

    const int b = blockIdx.x, tid = threadIdx.x, nt = blockDim.x;
    const float* predB = pred + (size_t)b * (NCC + 4) * AA;

    for (int w = tid; w < PAD_N / 32; w += nt) { sSupp[w] = 0u; sValid[w] = 0u; }
    __syncthreads();

    for (int r = tid; r < PAD_N; r += nt) {
        float4 nb = make_float4(0.f, 0.f, 0.f, 0.f);
        if (r < TOPK) {
            uint32_t a = topkIdx[(size_t)b * TOPK + r];
            uint32_t u = topkKey[(size_t)b * TOPK + r];
            float sc = keyToScore(u);
            float off = (float)cls[(size_t)b * AA + a] * MULT;
            nb.x = predB[(size_t)0 * AA + a] * BOX_SCALE + off;
            nb.y = predB[(size_t)1 * AA + a] * BOX_SCALE + off;
            nb.z = predB[(size_t)2 * AA + a] * BOX_SCALE + off;
            nb.w = predB[(size_t)3 * AA + a] * BOX_SCALE + off;
            if (sc > CONF_TH) atomicOr(&sValid[r >> 5], 1u << (r & 31));
        }
        sNms[r] = nb;
        sArea[r] = (nb.z - nb.x) * (nb.w - nb.y);
    }
    __syncthreads();

    const int wave = tid >> 5;
    const int lane = tid & 31;

    for (int g = 0; g < PAD_N / 64; ++g) {
        const int base = g * 64;

        bool mySupp = true, myValid = false;
        float4 myBox; float myAr = 0.0f;
        if (tid < 64) {
            mySupp  = ((sSupp [(base + tid) >> 5] >> ((base + tid) & 31)) & 1u) != 0u;
            myValid = ((sValid[(base + tid) >> 5] >> ((base + tid) & 31)) & 1u) != 0u;
            myBox   = sNms[base + tid];
            myAr    = sArea[base + tid];
        }
        for (int i = 0; i < 64; ++i) {
            if (tid == i) sKeep[i] = (!mySupp && myValid) ? 1 : 0;
            __syncthreads();
            if (tid < 64 && tid > i && sKeep[i]) {
                if (iouOver(sNms[base + i], sArea[base + i], myBox, myAr)) mySupp = true;
            }
        }
        __syncthreads();
        if (tid < 64 && mySupp)
            atomicOr(&sSupp[(base + tid) >> 5], 1u << ((base + tid) & 31));

        const int koff[8] = {0, 4, 8, 12, 32, 36, 40, 44};
        const int aHalf = (lane >> 4) << 4;
        v8i afrag;
#pragma unroll
        for (int v = 0; v < 8; ++v) {
            int k0 = koff[v] + aHalf;
            uint32_t w = (uint32_t)sKeep[k0]
                       | ((uint32_t)sKeep[k0 + 1] << 8)
                       | ((uint32_t)sKeep[k0 + 2] << 16)
                       | ((uint32_t)sKeep[k0 + 3] << 24);
            afrag[v] = (int)w;
        }
        const int firstTile = (base >> 4) + 4;
        for (int tIdx = firstTile + wave; tIdx < PAD_N / 16; tIdx += 32) {
            const int n = lane & 15;
            const int khalf = (lane >> 4) << 4;
            float4 colBox = sNms[tIdx * 16 + n];
            float  colAr  = sArea[tIdx * 16 + n];
            v8i bfrag;
#pragma unroll
            for (int v = 0; v < 8; ++v) {
                int kb = (v < 4) ? (khalf + v * 4) : (32 + khalf + (v - 4) * 4);
                uint32_t w = 0u;
#pragma unroll
                for (int e = 0; e < 4; ++e) {
                    bool hit = iouOver(sNms[base + kb + e], sArea[base + kb + e],
                                       colBox, colAr);
                    w |= (hit ? 1u : 0u) << (8 * e);
                }
                bfrag[v] = (int)w;
            }
            v8i cf = {};
            cf = __builtin_amdgcn_wmma_i32_16x16x64_iu8(false, afrag, false, bfrag, cf,
                                                        false, false);
            if (lane < 16 && cf[0] != 0) {
                int col = tIdx * 16 + lane;
                atomicOr(&sSupp[col >> 5], 1u << (col & 31));
            }
        }
        __syncthreads();
    }

    for (int w = tid; w < PAD_N / 32; w += nt) sKeepW[w] = (~sSupp[w]) & sValid[w];
    __syncthreads();
    if (tid == 0) {
        uint32_t run = 0;
        for (int w = 0; w < PAD_N / 32; ++w) { sPref[w] = run; run += (uint32_t)__popc(sKeepW[w]); }
        sPref[PAD_N / 32] = run;
    }
    __syncthreads();
    const uint32_t total = sPref[PAD_N / 32];
    float* outB = out + (size_t)b * MAX_DET * 6;

    for (int r = tid; r < MAX_DET; r += nt)
        if (r >= (int)total)
            for (int c = 0; c < 6; ++c) outB[r * 6 + c] = 0.0f;

    for (int r = tid; r < TOPK; r += nt) {
        uint32_t wrd = sKeepW[r >> 5];
        if ((wrd >> (r & 31)) & 1u) {
            uint32_t rank = sPref[r >> 5] + (uint32_t)__popc(wrd & ((1u << (r & 31)) - 1u));
            if (rank < MAX_DET) {
                uint32_t a = topkIdx[(size_t)b * TOPK + r];
                uint32_t u = topkKey[(size_t)b * TOPK + r];
                float* o = outB + (size_t)rank * 6;
                o[0] = predB[(size_t)0 * AA + a];
                o[1] = predB[(size_t)1 * AA + a];
                o[2] = predB[(size_t)2 * AA + a];
                o[3] = predB[(size_t)3 * AA + a];
                o[4] = keyToScore(u);
                o[5] = (float)cls[(size_t)b * AA + a];
            }
        }
    }
}

extern "C" void kernel_launch(void* const* d_in, const int* in_sizes, int n_in,
                              void* d_out, int out_size, void* d_ws, size_t ws_size,
                              hipStream_t stream) {
    (void)in_sizes; (void)n_in; (void)out_size; (void)ws_size;
    const float* pred = (const float*)d_in[0];
    float* out = (float*)d_out;

    uint32_t* keys    = (uint32_t*)d_ws;
    uint32_t* cls     = keys    + (size_t)BB * AA;
    uint32_t* topkKey = cls     + (size_t)BB * AA;
    uint32_t* topkIdx = topkKey + (size_t)BB * TOPK;

    dim3 g1((AA + 255) / 256, BB);
    score_kernel<<<g1, 256, 0, stream>>>(pred, keys, cls);
    topk_kernel<<<BB, 1024, 0, stream>>>(keys, topkKey, topkIdx);
    nms_kernel<<<BB, 1024, 0, stream>>>(pred, cls, topkKey, topkIdx, out);
}
